// SEG_HEAD_90623809946174
// MI455X (gfx1250) — matrix operand order repaired and checked by reading, not compiled, not run on hardware
//
#include <hip/hip_runtime.h>
#include <math.h>
#include <stdint.h>

// ---------------- problem constants ----------------
#define BATCH 2
#define NPT   4096
#define CH    128
#define MTOT  (BATCH * NPT)      // 8192
#define KNB   16                 // neighbors
#define NKP   43                 // kernel points
#define PCDIM (NKP * CH)         // 5504 (divisible by 32)
#define HID   (4 * CH)           // 512
#define CLS   16
#define RAD   0.1f
#define RAD2  0.01f

typedef __attribute__((ext_vector_type(16))) _Float16 v16h;
typedef __attribute__((ext_vector_type(8)))  float    v8f;
typedef __attribute__((ext_vector_type(4)))  unsigned v4u;
typedef __attribute__((ext_vector_type(8)))  unsigned v8u;

// ===================== WMMA fragment helpers (gfx1250 wave32) =====================
// 16-bit A matrix 16x32 (MxK), layout per CDNA5 ISA 7.12.2 (row-major LDS tile).
__device__ inline v16h load_a_frag_lds(const _Float16* tile, int stride) {
  int lane = threadIdx.x & 31;
  int row = lane & 15;
  int kb = (lane < 16) ? 0 : 8;
  v16h a;
#pragma unroll
  for (int h = 0; h < 16; ++h) {
    int vg = h >> 1, pos = h & 1;
    int k = ((vg & 4) ? 16 : 0) + kb + ((vg & 3) << 1) + pos;
    a[h] = tile[row * stride + k];
  }
  return a;
}

// ===================== Tensor Data Mover: 2D tile global->LDS =====================
// D# group0 (4 SGPRs) + group1 (8 SGPRs) per CDNA5 ISA.
// 2-byte elements, tile = tile_d0 x tile_d1, row stride = stride0 elements.
__device__ inline void tdm_load_2d(unsigned lds_off, const void* gaddr, unsigned stride0) {
  unsigned long long ga = (unsigned long long)(uintptr_t)gaddr;
  v4u g0;
  g0[0] = 1u;                                   // count=1 (valid user descriptor)
  g0[1] = lds_off;                              // lds_addr (bytes)
  g0[2] = (unsigned)(ga & 0xffffffffu);         // global_addr[31:0]
  g0[3] = (unsigned)((ga >> 32) & 0x01ffffffu)  // global_addr[56:32]
          | (2u << 30);                         // type=2 ("image")
  v8u g1;
  g1[0] = 1u << 16;                             // workgroup_mask=0, data_size=1 (2B)
  g1[1] = (0x7fffffffu & 0xffffu) << 16;        // tensor_dim0[15:0]
  g1[2] = (0x7fffffffu >> 16)                   // tensor_dim0[31:16]
          | ((0x7fffffffu & 0xffffu) << 16);    // tensor_dim1[15:0]
  g1[3] = (0x7fffffffu >> 16)                   // tensor_dim1[31:16]
          | (32u << 16);                        // tile_dim0 = 32 elements
  g1[4] = 16u;                                  // tile_dim1 = 16, tile_dim2 = 0
  g1[5] = stride0;                              // tensor_dim0_stride[31:0]
  g1[6] = 0u;                                   // stride0[47:32]=0, stride1 lo=0
  g1[7] = 0u;                                   // stride1 hi = 0
  asm volatile("tensor_load_to_lds %0, %1" :: "s"(g0), "s"(g1) : "memory");
}

// ===================== prep kernels =====================
__global__ void prep_xyz_kernel(const float* __restrict__ p, float* __restrict__ xyz) {
  int e = blockIdx.x * 256 + threadIdx.x;
  if (e >= MTOT * 3) return;
  int m = e / 3, d = e % 3;
  int b = m / NPT, n = m % NPT;
  xyz[e] = p[(size_t)b * 3 * NPT + (size_t)d * NPT + n];
}

__global__ void prep_xt_kernel(const float* __restrict__ x, float* __restrict__ xt) {
  int e = blockIdx.x * 256 + threadIdx.x; // MTOT*CH exact
  int m = e >> 7, c = e & 127;
  int b = m / NPT, n = m % NPT;
  xt[e] = x[(size_t)b * CH * NPT + (size_t)c * NPT + n];
}

__global__ void kpts_kernel(float* __restrict__ kp) {
  int i = threadIdx.x;
  if (i >= NKP) return;
  float X = 0.f, Y = 0.f, Z = 0.f;
  if (i >= 1) {
    int mc = (i < 15) ? 14 : 28;
    int j = (i < 15) ? (i - 1) : (i - 15);
    float scale = (i < 15) ? (RAD * 0.5f) : RAD;
    float ii = (float)j + 0.5f;
    float phi = acosf(1.f - 2.f * ii / (float)mc);
    float theta = 3.14159265358979323846f * (1.f + 2.23606797749978969f) * ii;
    X = cosf(theta) * sinf(phi) * scale;
    Y = sinf(theta) * sinf(phi) * scale;
    Z = cosf(phi) * scale;
  }
  kp[i * 3 + 0] = X; kp[i * 3 + 1] = Y; kp[i * 3 + 2] = Z;
}

// Shuffle a K x N f32 weight matrix into WMMA B-fragment-native f16 layout:
// per 32x16 (KxN) block: 32 lanes x 16 contiguous halves.
//   lane = (n%16) + 16*(((k%32)/8)&1), elem j = (k%8) + 8*((k%32)/16)
__global__ void prep_bfrag_kernel(const float* __restrict__ W, _Float16* __restrict__ o,
                                  int K, int N) {
  int e = blockIdx.x * 256 + threadIdx.x;
  if (e >= K * N) return;
  int k = e / N, n = e % N;
  int chunk = k >> 5, kl = k & 31;
  int ntile = n >> 4, nl = n & 15;
  int lane = nl + (((kl >> 3) & 1) << 4);
  int j = (kl & 7) + ((kl & 16) ? 8 : 0);
  size_t off = ((size_t)chunk * (N >> 4) + ntile) * 512 + (size_t)lane * 16 + j;
  o[off] = (_Float16)W[e];
}

// ===================== KNN (wave32 per point) =====================
__global__ __launch_bounds__(32) void knn_kernel(const float* __restrict__ xyz,
                                                 int* __restrict__ idx_out) {
  int m = blockIdx.x;
  int b = m / NPT;
  int lane = threadIdx.x;
  float px = xyz[m * 3 + 0], py = xyz[m * 3 + 1], pz = xyz[m * 3 + 2];
  float dl[KNB]; int il[KNB];
#pragma unroll
  for (int i = 0; i < KNB; ++i) { dl[i] = 3.4e38f; il[i] = m; }
  int base = b * NPT;
  for (int c = lane; c < NPT; c += 32) {
    int g = base + c;
    float dx = xyz[g * 3 + 0] - px, dy = xyz[g * 3 + 1] - py, dz = xyz[g * 3 + 2] - pz;
    float d = dx * dx + dy * dy + dz * dz;
    if (d < dl[KNB - 1]) {
      int pos = KNB - 1;
      while (pos > 0 && dl[pos - 1] > d) { dl[pos] = dl[pos - 1]; il[pos] = il[pos - 1]; --pos; }
      dl[pos] = d; il[pos] = g;
    }
  }
  __shared__ float sd[KNB];
  __shared__ int   si[KNB];
  int ptr = 0;
#pragma unroll
  for (int r = 0; r < KNB; ++r) {
    float v  = (ptr < KNB) ? dl[ptr] : 3.4e38f;
    int cand = (ptr < KNB) ? il[ptr] : m;
    float bv = v; int bl = lane;
    for (int off = 16; off > 0; off >>= 1) {
      float ov = __shfl_xor(bv, off, 32);
      int   ol = __shfl_xor(bl, off, 32);
      if (ov < bv || (ov == bv && ol < bl)) { bv = ov; bl = ol; }
    }
    int widx = __shfl(cand, bl, 32);
    if (lane == bl && ptr < KNB) ++ptr;
    if (lane == 0) { sd[r] = bv; si[r] = widx; }
  }
  __syncthreads();
  if (lane < KNB) {
    int sel = (sd[lane] <= RAD2) ? si[lane] : si[0];
    idx_out[(size_t)m * KNB + lane] = sel;
  }
}

// ===================== fused KPConv (WMMA) =====================
// out[m,d] = fin[m,d] + sum_{p,c} fk[m,p,c] * W[p*CH+c, d]
// fk[m,p,c] = sum_k infl[m,k,p] * fin[idx[m,k], c]
// Whf is in B-fragment-native layout (prep_bfrag_kernel, K=PCDIM, N=CH).
__global__ __launch_bounds__(256) void kpconv_kernel(const float* __restrict__ xyz,
                                                     const int* __restrict__ idx,
                                                     const float* __restrict__ fin,
                                                     const _Float16* __restrict__ Whf,
                                                     const float* __restrict__ kpts,
                                                     float* __restrict__ fout) {
  __shared__ _Float16 s_feat[16][KNB][CH];        // 64 KB
  __shared__ _Float16 s_infl[16][KNB][NKP + 1];   // 22 KB
  __shared__ float    s_rel[16][KNB][3];
  __shared__ int      s_idx[16][KNB];
  __shared__ float    s_kp[NKP][3];
  __shared__ __align__(128) _Float16 s_a[512];    // A tile in fragment order
  int m0 = blockIdx.x * 16;
  int tid = threadIdx.x;

  if (tid < NKP * 3) s_kp[tid / 3][tid % 3] = kpts[tid];
  for (int e = tid; e < 16 * KNB; e += 256) {
    int pt = e >> 4, k = e & 15;
    int g = idx[(size_t)(m0 + pt) * KNB + k];
    s_idx[pt][k] = g;
    int mm = m0 + pt;
    s_rel[pt][k][0] = xyz[g * 3 + 0] - xyz[mm * 3 + 0];
    s_rel[pt][k][1] = xyz[g * 3 + 1] - xyz[mm * 3 + 1];
    s_rel[pt][k][2] = xyz[g * 3 + 2] - xyz[mm * 3 + 2];
  }
  __syncthreads();

  for (int e = tid; e < 16 * KNB * CH; e += 256) {
    int pt = e >> 11;
    int rem = e & 2047;
    int k = rem >> 7, c = rem & 127;
    s_feat[pt][k][c] = (_Float16)fin[(size_t)s_idx[pt][k] * CH + c];
  }
  for (int e = tid; e < 16 * KNB * NKP; e += 256) {
    int pt = e / (KNB * NKP);
    int rem = e % (KNB * NKP);
    int k = rem / NKP, pp = rem % NKP;
    float dx = s_rel[pt][k][0] - s_kp[pp][0];
    float dy = s_rel[pt][k][1] - s_kp[pp][1];
    float dz = s_rel[pt][k][2] - s_kp[pp][2];
    float d = sqrtf(dx * dx + dy * dy + dz * dz + 1e-12f);
    float v = 1.0f - d * (1.0f / RAD);
    s_infl[pt][k][pp] = (_Float16)(v > 0.f ? v : 0.f);
  }
  __syncthreads();

  int wave = tid >> 5, lane = tid & 31;
  v8f acc = {};
  int r = tid >> 4;             // A-build: row for this thread
  int j0 = (tid & 15) << 1;     // two adjacent k columns
  for (int t = 0; t < PCDIM / 32; ++t) {   // 172 chunks
    int p = t >> 2, c0 = (t & 3) * 32;
    // cooperatively build A tile in fragment order
    float s0 = 0.f, s1 = 0.f;
#pragma unroll
    for (int k = 0; k < KNB; ++k) {
      float w = (float)s_infl[r][k][p];
      s0 += w * (float)s_feat[r][k][c0 + j0];
      s1 += w * (float)s_feat[r][k][c0 + j0 + 1];
    }
    {
      int ln = r + (((j0 >> 3) & 1) << 4);
      int h = (j0 & 7) + ((j0 & 16) ? 8 : 0);
      s_a[ln * 16 + h] = (_Float16)s0;
      s_a[ln * 16 + h + 1] = (_Float16)s1;   // j0 even => same lane, h+1
    }
    __syncthreads();
    v16h a = *(const v16h*)(s_a + lane * 16);
    const _Float16* bp = Whf + ((size_t)t * 8 + wave) * 512 + (size_t)lane * 16;
    v16h b = *(const v16h*)bp;
    if (t + 1 < PCDIM / 32)
      __builtin_prefetch(Whf + ((size_t)(t + 1) * 8 + wave) * 512 + (size_t)lane * 16, 0, 1);
    acc = __builtin_amdgcn_wmma_f32_16x16x32_f16(false, a, false, b, (short)0, acc, false, false);
    __syncthreads();
  }
  int col = wave * 16 + (lane & 15);
  int rb = (lane >> 4) * 8;
#pragma unroll
  for (int rr = 0; rr < 8; ++rr) {
    int m = m0 + rb + rr;
    size_t o = (size_t)m * CH + col;
    fout[o] = acc[rr] + fin[o];
  }
}

// ===================== generic WMMA GEMM with TDM A-staging ======================
// C[MxN] = A[MxK](f16, row-major) * B (f16, fragment-native layout) (+addsrc)
__global__ __launch_bounds__(256) void gemm_wmma_kernel(const _Float16* __restrict__ A,
                                                        const _Float16* __restrict__ Bf,
                                                        float* __restrict__ C,
                                                        const float* __restrict__ addsrc,
                                                        int M, int N, int K) {
  __shared__ __align__(128) _Float16 At[16 * 32];
  int m0 = blockIdx.x * 16;
  int wave = threadIdx.x >> 5, lane = threadIdx.x & 31;
  int ntile = blockIdx.y * 8 + wave;
  int ntiles = N >> 4;
  unsigned lds_off = (unsigned)(uintptr_t)&At[0];
  v8f acc = {};
  int nchunks = K >> 5;
  for (int t = 0; t < nchunks; ++t) {
    __syncthreads();                       // everyone done with previous tile
    if (threadIdx.x < 32) {                // wave 0 drives the Tensor Data Mover
      tdm_load_2d(lds_off, A + (size_t)m0 * K + t * 32, (unsigned)K);
      __builtin_amdgcn_s_wait_tensorcnt(0);
    }
    __syncthreads();                       // tile visible to all waves
    v16h a = load_a_frag_lds(At, 32);
    const _Float16* bp = Bf + ((size_t)t * ntiles + ntile) * 512 + (size_t)lane * 16;
    v16h b = *(const v16h*)bp;
    if (t + 1 < nchunks)
      __builtin_prefetch(Bf + ((size_t)(t + 1) * ntiles + ntile) * 512 + (size_t)lane * 16, 0, 1);
    acc = __builtin_amdgcn_wmma_f32_16x16x32_f16(false, a, false, b, (short)0, acc, false, false);
  }
  int col = (blockIdx.y * 8 + wave) * 16 + (lane & 15);
  int rb = (lane >> 4) * 8;
#pragma unroll
  for (int r = 0; r < 8; ++r) {
    int m = m0 + rb + r;
    size_t o = (size_t)m * N + col;
    float v = acc[r];
    if (addsrc) v += addsrc[o];
    C[o] = v;
  }
}

// ===================== LayerNorm over channel dim =====================
__global__ __launch_bounds__(128) void ln_kernel(const float* __restrict__ x,
                                                 const float* __restrict__ g,
                                                 const float* __restrict__ b,
                                                 float* __restrict__ y) {
  int m = blockIdx.x, c = threadIdx.x;
  __shared__ float red[CH];
  float v = x[(size_t)m * CH + c];
  red[c] = v; __syncthreads();
  for (int s = 64; s > 0; s >>= 1) { if (c < s) red[c] += red[c + s]; __syncthreads(); }
  float mu = red[0] * (1.0f / CH);
  __syncthreads();
  float d = v - mu;
  red[c] = d * d; __syncthreads();
  for (int s = 64; s > 0; s >>= 1) { if (c < s) red[c] += red[c + s]; __syncthreads(); }
  float rs = rsqrtf(red[0] * (1.0f / CH) + 1e-5f);
  y[(size_t)m * CH + c] = d * rs * g[c] + b[c];
}

// ===================== token-shift mixes =====================
__global__ void mix3_kernel(const float* __restrict__ xn,
                            const float* __restrict__ mk, const float* __restrict__ mv,
                            const float* __restrict__ mr,
                            _Float16* __restrict__ ok, _Float16* __restrict__ ov,
                            _Float16* __restrict__ orr) {
  int e = blockIdx.x * 256 + threadIdx.x; // MTOT*CH exact
  int m = e >> 7, c = e & 127;
  float xv = xn[e];
  float xs = (m % NPT == 0) ? 0.f : xn[e - CH];
  float a = mk[c]; ok[e]  = (_Float16)(xv * a + xs * (1.f - a));
  a = mv[c];       ov[e]  = (_Float16)(xv * a + xs * (1.f - a));
  a = mr[c];       orr[e] = (_Float16)(xv * a + xs * (1.f - a));
}

__global__ void mix2_kernel(const float* __restrict__ xn,
                            const float* __restrict__ mk, const float* __restrict__ mr,
                            _Float16* __restrict__ ok, _Float16* __restrict__ orr) {
  int e = blockIdx.x * 256 + threadIdx.x;
  int m = e >> 7, c = e & 127;
  float xv = xn[e];
  float xs = (m % NPT == 0) ? 0.f : xn[e - CH];
  float a = mk[c]; ok[e]  = (_Float16)(xv * a + xs * (1.f - a));
  a = mr[c];       orr[e] = (_Float16)(xv * a + xs * (1.f - a));
}

// ===================== WKV recurrence (sequential per (b,c)) =====================
__global__ __launch_bounds__(256) void wkv_kernel(const float* __restrict__ kk,
                                                  const float* __restrict__ vv,
                                                  const float* __restrict__ td,
                                                  const float* __restrict__ tfirst,
                                                  float* __restrict__ out) {
  int t = threadIdx.x;          // 256 = BATCH*CH
  int b = t >> 7, c = t & 127;
  float w = -expf(td[c]);
  float tfv = tfirst[c];
  float aa = 0.f, bb = 0.f, pp = -1e38f;
  size_t base = (size_t)b * NPT * CH + c;
  for (int n = 0; n < NPT; ++n) {
    size_t o = base + (size_t)n * CH;
    float kt = kk[o], vt = vv[o];
    float ww = tfv + kt;
    float q = fmaxf(pp, ww);
    float e1 = expf(pp - q), e2 = expf(ww - q);
    out[o] = (e1 * aa + e2 * vt) / (e1 * bb + e2);
    float ww2 = pp + w;
    float q2 = fmaxf(ww2, kt);
    e1 = expf(ww2 - q2); e2 = expf(kt - q2);
    aa = e1 * aa + e2 * vt;
    bb = e1 * bb + e2;
    pp = q2;
  }
}

// ===================== elementwise =====================
__global__ void add2_kernel(const float* a, const float* b, float* o, int n) {
  int e = blockIdx.x * 256 + threadIdx.x;
  if (e < n) o[e] = a[e] + b[e];
}
__global__ void sigmul16_kernel(const float* r, const float* w, _Float16* o, int n) {
  int e = blockIdx.x * 256 + threadIdx.x;
  if (e < n) {
    float s = 1.f / (1.f + expf(-r[e]));
    o[e] = (_Float16)(s * w[e]);
  }
}
__global__ void relusq16_kernel(const float* a, _Float16* o, int n) {
  int e = blockIdx.x * 256 + threadIdx.x;
  if (e < n) {
    float v = a[e]; v = v > 0.f ? v : 0.f;
    o[e] = (_Float16)(v * v);
  }
}
__global__ void ffnout_kernel(const float* base, const float* r, const float* kv, float* o, int n) {
  int e = blockIdx.x * 256 + threadIdx.x;
  if (e < n) {
    float s = 1.f / (1.f + expf(-r[e]));
    o[e] = base[e] + s * kv[e];
  }
}

// ===================== output head =====================
__global__ void yout_kernel(const float* __restrict__ pts, float* __restrict__ out) {
  int e = blockIdx.x * 256 + threadIdx.x; // MTOT*CH exact
  int b = e >> 19;
  int rem = e & 524287;
  int c = rem >> 12, n = rem & 4095;
  out[e] = pts[((size_t)(b * NPT + n)) * CH + c];
}

__global__ __launch_bounds__(256) void bn_stats_kernel(const float* __restrict__ x,
                                                       float* __restrict__ mu,
                                                       float* __restrict__ rs) {
  int c = blockIdx.x, t = threadIdx.x;
  float s = 0.f, s2 = 0.f;
  for (int m = t; m < MTOT; m += 256) {
    float v = x[(size_t)m * CH + c];
    s += v; s2 += v * v;
  }
  __shared__ float r1[256], r2[256];
  r1[t] = s; r2[t] = s2; __syncthreads();
  for (int k = 128; k > 0; k >>= 1) {
    if (t < k) { r1[t] += r1[t + k]; r2[t] += r2[t + k]; }
    __syncthreads();
  }
  if (t == 0) {
    float mean = r1[0] * (1.0f / MTOT);
    float var = r2[0] * (1.0f / MTOT) - mean * mean;
    mu[c] = mean;
    rs[c] = rsqrtf(var + 1e-5f);
  }
}

__global__ __launch_bounds__(128) void label_kernel(const float* __restrict__ pts,
                                                    const float* __restrict__ mu,
                                                    const float* __restrict__ rsb,
                                                    const float* __restrict__ g,
                                                    const float* __restrict__ bb,
                                                    const float* __restrict__ cw,
                                                    const float* __restrict__ cb,
                                                    float* __restrict__ out) {
  int m = blockIdx.x, c = threadIdx.x;
  __shared__ float yn[CH];
  float v = pts[(size_t)m * CH + c];
  float t = (v - mu[c]) * rsb[c] * g[c] + bb[c];
  yn[c] = t > 0.f ? t : 0.f;
  __syncthreads();
  if (c < CLS) {
    float s = cb[c];
    for (int j = 0; j < CH; ++j) s += yn[j] * cw[c * CH + j];
    int b = m / NPT, n = m % NPT;
    out[(size_t)b * CLS * NPT + (size_t)c * NPT + n] = s;
  }
}

// ===================== host-side launch =====================
extern "C" void kernel_launch(void* const* d_in, const int* in_sizes, int n_in,
                              void* d_out, int out_size, void* d_ws, size_t ws_size,
                              hipStream_t stream) {
  (void)in_sizes; (void)n_in; (void)out_size; (void)ws_size;
  const float* p    = (const float*)d_in[0];
  const float* x    = (const float*)d_in[1];
  const float* kp1w = (const float*)d_in[2];
  const float* kp2w = (const float*)d_in[3];
  const float* ln1g = (const float*)d_in[4];
  const float* ln1b = (const float*)d_in[5];
  const float* td   = (const float*)d_in[6];
  const float* tfv  = (const float*)d_in[7];
  const float* mixk = (const float*)d_in[8];
  const float* mixv = (const float*)d_in[9];
  const float* mixr = (const float*)d_in[10];
  const float* awk  = (const float*)d_in[11];
  const float* awv  = (const float*)d_in[12];
  const float* awr  = (const float*)d_in[13];
  const float* awo  = (const float*)d_in[14];
  const float* ln2g = (const float*)d_in[15];
  const float* ln2b = (const float*)d_in[16];
  const float* cmk  = (const float*)d_in[17];
  const float* cmr  = (const float*)d_in[18];
  const float* fwk  = (const float*)d_in[19];
  const float* fwv  = (const float*)d_in[20];
  const float* fwr  = (const float*)d_in[21];
  const float* bng  = (const float*)d_in[22];
  const float* bnb  = (const float*)d_in[23];
  const float* cw   = (const float*)d_in[24];
  const float* cb   = (const float*)d_in[25];
  float* out = (float*)d_out;

  // ---- deterministic workspace carve ----
  char* wp = (char*)d_ws;
  auto alloc = [&](size_t bytes) -> void* {
    void* r = (void*)wp;
    wp += (bytes + 255) & ~(size_t)255;
    return r;
  };
  const size_t MC = (size_t)MTOT * CH;
  float*    xyz   = (float*)alloc((size_t)MTOT * 3 * 4);
  float*    kpbuf = (float*)alloc(NKP * 3 * 4);
  int*      idxb  = (int*)alloc((size_t)MTOT * KNB * 4);
  float*    xt    = (float*)alloc(MC * 4);
  float*    fA    = (float*)alloc(MC * 4);
  float*    fB    = (float*)alloc(MC * 4);
  _Float16* w1h   = (_Float16*)alloc((size_t)PCDIM * CH * 2);
  _Float16* w2h   = (_Float16*)alloc((size_t)PCDIM * CH * 2);
  _Float16* wkh   = (_Float16*)alloc((size_t)CH * CH * 2);
  _Float16* wvh   = (_Float16*)alloc((size_t)CH * CH * 2);
  _Float16* wrh   = (_Float16*)alloc((size_t)CH * CH * 2);
  _Float16* woh   = (_Float16*)alloc((size_t)CH * CH * 2);
  _Float16* fwkh  = (_Float16*)alloc((size_t)CH * HID * 2);
  _Float16* fwvh  = (_Float16*)alloc((size_t)HID * CH * 2);
  _Float16* fwrh  = (_Float16*)alloc((size_t)CH * CH * 2);
  float*    xn    = (float*)alloc(MC * 4);
  _Float16* mk16  = (_Float16*)alloc(MC * 2);
  _Float16* mv16  = (_Float16*)alloc(MC * 2);
  _Float16* mr16  = (_Float16*)alloc(MC * 2);
  float*    kkb   = (float*)alloc(MC * 4);
  float*    vvb   = (float*)alloc(MC * 4);
  float*    rrb   = (float*)alloc(MC * 4);
  float*    wkvb  = (float*)alloc(MC * 4);
  _Float16* rw16  = (_Float16*)alloc(MC * 2);
  float*    hid   = (float*)alloc((size_t)MTOT * HID * 4);
  _Float16* hid16 = (_Float16*)alloc((size_t)MTOT * HID * 2);
  float*    bnmu  = (float*)alloc(CH * 4);
  float*    bnrs  = (float*)alloc(CH * 4);

  const int EW = 256;
  auto blks = [](size_t n) { return (unsigned)((n + 255) / 256); };

  // ---- prep ----
  prep_xyz_kernel<<<blks(MTOT * 3), EW, 0, stream>>>(p, xyz);
  prep_xt_kernel<<<blks(MC), EW, 0, stream>>>(x, xt);
  kpts_kernel<<<1, 64, 0, stream>>>(kpbuf);
  knn_kernel<<<MTOT, 32, 0, stream>>>(xyz, idxb);
  prep_bfrag_kernel<<<blks((size_t)PCDIM * CH), EW, 0, stream>>>(kp1w, w1h, PCDIM, CH);
  prep_bfrag_kernel<<<blks((size_t)PCDIM * CH), EW, 0, stream>>>(kp2w, w2h, PCDIM, CH);
  prep_bfrag_kernel<<<blks(CH * CH), EW, 0, stream>>>(awk, wkh, CH, CH);
  prep_bfrag_kernel<<<blks(CH * CH), EW, 0, stream>>>(awv, wvh, CH, CH);
  prep_bfrag_kernel<<<blks(CH * CH), EW, 0, stream>>>(awr, wrh, CH, CH);
  prep_bfrag_kernel<<<blks(CH * CH), EW, 0, stream>>>(awo, woh, CH, CH);
  prep_bfrag_kernel<<<blks(CH * HID), EW, 0, stream>>>(fwk, fwkh, CH, HID);
  prep_bfrag_kernel<<<blks(HID * CH), EW, 0, stream>>>(fwv, fwvh, HID, CH);
  prep_bfrag_kernel<<<blks(CH * CH), EW, 0, stream>>>(fwr, fwrh, CH, CH);

  // ---- two KPConv blocks (WMMA) with residuals ----
  kpconv_kernel<<<MTOT / 16, 256, 0, stream>>>(xyz, idxb, xt, w1h, kpbuf, fA);
  kpconv_kernel<<<MTOT / 16, 256, 0, stream>>>(xyz, idxb, fA, w2h, kpbuf, fB);
  add2_kernel<<<blks(MC), EW, 0, stream>>>(fB, xt, fA, (int)MC);   // pts in fA

  // ---- RWKV spatial mix ----
  ln_kernel<<<MTOT, CH, 0, stream>>>(fA, ln1g, ln1b, xn);
  mix3_kernel<<<blks(MC), EW, 0, stream>>>(xn, mixk, mixv, mixr, mk16, mv16, mr16);
  gemm_wmma_kernel<<<dim3(MTOT / 16, 1), 256, 0, stream>>>(mk16, wkh, kkb, nullptr, MTOT, CH, CH);
  gemm_wmma_kernel<<<dim3(MTOT / 16, 1), 256, 0, stream>>>(mv16, wvh, vvb, nullptr, MTOT, CH, CH);
  gemm_wmma_kernel<<<dim3(MTOT / 16, 1), 256, 0, stream>>>(mr16, wrh, rrb, nullptr, MTOT, CH, CH);
  wkv_kernel<<<1, 256, 0, stream>>>(kkb, vvb, td, tfv, wkvb);
  sigmul16_kernel<<<blks(MC), EW, 0, stream>>>(rrb, wkvb, rw16, (int)MC);
  gemm_wmma_kernel<<<dim3(MTOT / 16, 1), 256, 0, stream>>>(rw16, woh, fB, fA, MTOT, CH, CH); // pts2 in fB

  // ---- channel mix ----
  ln_kernel<<<MTOT, CH, 0, stream>>>(fB, ln2g, ln2b, xn);
  mix2_kernel<<<blks(MC), EW, 0, stream>>>(xn, cmk, cmr, mk16, mr16);
  gemm_wmma_kernel<<<dim3(MTOT / 16, HID / 128), 256, 0, stream>>>(mk16, fwkh, hid, nullptr, MTOT, HID, CH);
  relusq16_kernel<<<blks((size_t)MTOT * HID), EW, 0, stream>>>(hid, hid16, MTOT * HID);
  gemm_wmma_kernel<<<dim3(MTOT / 16, 1), 256, 0, stream>>>(hid16, fwvh, kkb, nullptr, MTOT, CH, HID);
  gemm_wmma_kernel<<<dim3(MTOT / 16, 1), 256, 0, stream>>>(mr16, fwrh, rrb, nullptr, MTOT, CH, CH);
  ffnout_kernel<<<blks(MC), EW, 0, stream>>>(fB, rrb, kkb, fA, (int)MC);  // pts3 in fA

  // ---- outputs: y then label ----
  yout_kernel<<<blks(MC), EW, 0, stream>>>(fA, out);
  bn_stats_kernel<<<CH, 256, 0, stream>>>(fA, bnmu, bnrs);
  label_kernel<<<MTOT, CH, 0, stream>>>(fA, bnmu, bnrs, bng, bnb, cw, cb, out + MC);
}
